// Block_63282048139763
// MI455X (gfx1250) — matrix operand order repaired and checked by reading, not compiled, not run on hardware
//
#include <hip/hip_runtime.h>
#include <hip/hip_bf16.h>

// RWKV-7 chunked scan for MI455X (gfx1250), bf16 WMMA with f32 accumulation,
// double-buffered GLOBAL_LOAD_ASYNC_TO_LDS_B128 input pipeline (ASYNCcnt).
// B=2, T=4096, H=32, C=64, DT=16. One workgroup per (b,h) head, 4 waves.

typedef __attribute__((ext_vector_type(16))) __bf16 v16bf;
typedef __attribute__((ext_vector_type(8)))  float  v8f;

#define N_HEADS 64
#define H_PER_B 32
#define CH      64
#define HC      2048
#define T_TOT   4096
#define DTC     16
#define NCHUNK  256

union V16U { v16bf v; unsigned u[8]; };

// A operand (16xK tile starting at row 0, col k0) from row-major bf16 [m][ld].
// 16-bit A layout: lane m = l%16; VGPR i, pack p -> K = (i&3)*2 + p + (l/16)*8 + (i/4)*16.
static __device__ __forceinline__ v16bf ldsA(const __bf16* p, int ld, int k0, int lane) {
  const int m = lane & 15, h = lane >> 4;
  V16U r;
#pragma unroll
  for (int i = 0; i < 8; ++i) {
    const int K = k0 + ((i & 3) << 1) + (h << 3) + ((i >> 2) << 4);
    r.u[i] = *(const unsigned*)(p + m * ld + K);
  }
  return r.v;
}

// B operand (Kx16 tile, cols n0..n0+15, rows k0..) from N-major bf16 [n][ld] (ld over K).
// 16-bit B layout: lane n = n0 + l%16; VGPR i, pack p -> K = (i&3)*2 + p + (l/16)*8 + (i/4)*16.
static __device__ __forceinline__ v16bf ldsB(const __bf16* p, int ld, int n0, int k0, int lane) {
  const int n = n0 + (lane & 15), h = lane >> 4;
  V16U r;
#pragma unroll
  for (int i = 0; i < 8; ++i) {
    const int K = k0 + ((i & 3) << 1) + (h << 3) + ((i >> 2) << 4);
    r.u[i] = *(const unsigned*)(p + n * ld + K);
  }
  return r.v;
}

static __device__ __forceinline__ v8f wmma_bf16(v16bf a, v16bf b, v8f c) {
  return __builtin_amdgcn_wmma_f32_16x16x32_bf16(false, a, false, b, (short)0, c, false, false);
}

__global__ __launch_bounds__(128, 1)
void rwkv7_heads_kernel(const float* __restrict__ gw, const float* __restrict__ gq,
                        const float* __restrict__ gk, const float* __restrict__ gv,
                        const float* __restrict__ ga, const float* __restrict__ gb,
                        float* __restrict__ gy) {
  // bf16 staging (row-major over last index)
  __shared__ __align__(16) __bf16 waA[DTC * CH];   // A: wa[t][c]
  __shared__ __align__(16) __bf16 wqA[DTC * CH];   // A: wq[t][c]
  __shared__ __align__(16) __bf16 kwiN[DTC * CH];  // B (n=s,k=c): kwi[s][c]
  __shared__ __align__(16) __bf16 bwiN[DTC * CH];  // B (n=s,k=c): bwi[s][c]
  __shared__ __align__(16) __bf16 Sb[CH * CH];     // B (n=i,k=j): state mirror S[i][j]
  __shared__ __align__(16) __bf16 vuB[CH * 32];    // B (n=i): k<16 -> v[t][i], k>=16 -> u[t][i]
  __shared__ __align__(16) __bf16 kbT[CH * 32];    // A (m=j): k<16 -> kwi[t][j]*fw[j], k>=16 -> bwi[t][j]*fw[j]
  __shared__ __align__(16) __bf16 qkqbA[DTC * 32]; // A: k<16 -> qk (m2), k>=16 -> qb (m2)
  __shared__ __align__(16) __bf16 akA[DTC * 32];   // A: k<16 -> ak (m1), k>=16 -> 0
  __shared__ __align__(16) float  abf[DTC * DTC];  // f32 strict-lower ab for the triangular solve
  __shared__ __align__(16) float  uf[DTC * CH];    // f32 ab_u / u workspace
  __shared__ __align__(16) float  fwl[CH];         // f32 per-channel chunk decay fw[j]
  // double-buffered raw f32 input staging, filled by async DMA:
  // rawf[buf][tensor*1024 + t*64 + c], tensor order: w,q,k,v,a,b
  __shared__ __align__(16) float  rawf[2][6 * DTC * CH];

  const int tid  = threadIdx.x;
  const int lane = tid & 31;
  const int wave = tid >> 5;
  const int head = blockIdx.x;                 // 0..63
  const int bidx = head >> 5;                  // / H
  const int hh   = head & (H_PER_B - 1);
  const size_t base0 = (size_t)bidx * T_TOT * HC + (size_t)hh * CH;

  const int i0   = wave << 4;                  // this wave's output/state column tile
  const int n_lo = lane & 15;
  const int hlf  = lane >> 4;

  // --- async issue of one chunk's raw inputs into rawf[buf] ------------------
  // 1536 B128 transfers per chunk; 12 per thread; tensor index uniform per step.
  auto issue_chunk = [&](int chunk, int buf) {
    const size_t tb = base0 + (size_t)chunk * DTC * HC;
    const unsigned rawoff = (unsigned)(uintptr_t)(const void*)&rawf[buf][0];
#pragma unroll
    for (int j = 0; j < 12; ++j) {
      const float* g0 = (j < 2) ? gw : (j < 4) ? gq : (j < 6) ? gk
                       : (j < 8) ? gv : (j < 10) ? ga : gb;       // uniform per j
      const int rem = tid + ((j & 1) << 7);                        // 0..255
      const int t = rem >> 4, c16 = rem & 15;                      // row, 16-float block
      const float* gp = g0 + tb + (size_t)t * HC + (c16 << 4);
      const unsigned loff =
          rawoff + (unsigned)((((j >> 1) * DTC * CH) + t * CH + (c16 << 4)) * 4);
      asm volatile("global_load_async_to_lds_b128 %0, %1, off"
                   :: "v"(loff), "v"((unsigned long long)(uintptr_t)gp)
                   : "memory");
    }
  };

  // zero-init: state mirror, akA zero pad, vuB (u-half read-before-write in chunk 0)
  issue_chunk(0, 0);                           // start DMA for chunk 0 ASAP
  for (int i = tid; i < CH * CH; i += 128) Sb[i] = (__bf16)0.f;
  for (int i = tid; i < DTC * 32; i += 128) akA[i] = (__bf16)0.f;
  for (int i = tid; i < CH * 32; i += 128) vuB[i] = (__bf16)0.f;
  v8f st[4];
#pragma unroll
  for (int jt = 0; jt < 4; ++jt) st[jt] = (v8f){0.f, 0.f, 0.f, 0.f, 0.f, 0.f, 0.f, 0.f};

  for (int nc = 0; nc < NCHUNK; ++nc) {
    const int buf = nc & 1;
    const size_t tbase = base0 + (size_t)nc * DTC * HC;
    const float* raw = &rawf[buf][0];

    // Kick off next chunk's DMA (its target buffer was last read >=1 barrier ago),
    // then wait for the current chunk's 12 in-order async loads to complete.
    if (nc + 1 < NCHUNK) {
      issue_chunk(nc + 1, buf ^ 1);
      asm volatile("s_wait_asynccnt 0xc" ::: "memory");   // <=12: current chunk landed
    } else {
      asm volatile("s_wait_asynccnt 0x0" ::: "memory");
    }
    __syncthreads();  // make freshly-landed raw[] visible to all waves

    // ---- phase 1: per-channel elementwise + cumprod decay chain (from LDS) ----
    if (tid < CH) {
      const int c = tid;
      float incl = 1.f;
      float kwr[DTC], bwr[DTC];
#pragma unroll
      for (int t = 0; t < DTC; ++t) {
        const int ro = t * CH + c;
        const float wv = raw[0 * DTC * CH + ro], qv = raw[1 * DTC * CH + ro];
        const float kv = raw[2 * DTC * CH + ro], vv = raw[3 * DTC * CH + ro];
        const float av = raw[4 * DTC * CH + ro], bv = raw[5 * DTC * CH + ro];
        const float dec = __expf(-__expf(wv));
        const float noninc = incl;       // exclusive cumprod
        incl *= dec;                     // inclusive cumprod
        const float inv = 1.f / incl;
        waA[t * CH + c]  = (__bf16)(av * noninc);
        wqA[t * CH + c]  = (__bf16)(qv * incl);
        kwiN[t * CH + c] = (__bf16)(kv * inv);
        bwiN[t * CH + c] = (__bf16)(bv * inv);
        vuB[c * 32 + t]  = (__bf16)vv;   // transposed: v[t][i=c]
        kwr[t] = kv * inv;
        bwr[t] = bv * inv;
      }
      fwl[c] = incl;                     // fw[c] = full-chunk decay
#pragma unroll
      for (int t = 0; t < DTC; ++t) {
        kbT[c * 32 + t]      = (__bf16)(kwr[t] * incl);
        kbT[c * 32 + 16 + t] = (__bf16)(bwr[t] * incl);
      }
    }
    __syncthreads();

    // ---- phase 2: intra-chunk 16x16 matrices, one per wave ----
    // wave0: ab = wa@bwi^T (m1, f32)   wave1: ak = wa@kwi^T (m1, bf16 A)
    // wave2: qk = wq@kwi^T (m2)        wave3: qb = wq@bwi^T (m2)
    {
      const __bf16* Ap = (wave < 2) ? waA : wqA;
      const __bf16* Bp = (wave == 0 || wave == 3) ? bwiN : kwiN;
      v8f d = (v8f){0.f, 0.f, 0.f, 0.f, 0.f, 0.f, 0.f, 0.f};
      d = wmma_bf16(ldsA(Ap, CH, 0, lane),  ldsB(Bp, CH, 0, 0, lane),  d);
      d = wmma_bf16(ldsA(Ap, CH, 32, lane), ldsB(Bp, CH, 0, 32, lane), d);
#pragma unroll
      for (int r = 0; r < 8; ++r) {
        const int t = r + (hlf << 3);
        const int s = n_lo;
        const bool keep = (wave < 2) ? (t > s) : (t >= s);
        const float val = keep ? d[r] : 0.f;
        if (wave == 0)      abf[t * DTC + s]       = val;
        else if (wave == 1) akA[t * 32 + s]        = (__bf16)val;
        else if (wave == 2) qkqbA[t * 32 + s]      = (__bf16)val;
        else                qkqbA[t * 32 + 16 + s] = (__bf16)val;
      }
    }
    __syncthreads();

    // ---- phase 3: ab_u = ak@v + wa@S^T  (this wave's 16-col i-tile) ----
    {
      v8f d = (v8f){0.f, 0.f, 0.f, 0.f, 0.f, 0.f, 0.f, 0.f};
      d = wmma_bf16(ldsA(akA, 32, 0, lane), ldsB(vuB, 32, i0, 0, lane), d); // zeros kill u-half
      d = wmma_bf16(ldsA(waA, CH, 0, lane),  ldsB(Sb, CH, i0, 0, lane),  d);
      d = wmma_bf16(ldsA(waA, CH, 32, lane), ldsB(Sb, CH, i0, 32, lane), d);
#pragma unroll
      for (int r = 0; r < 8; ++r) uf[(r + (hlf << 3)) * CH + i0 + n_lo] = d[r];
    }
    __syncthreads();

    // ---- phase 4: unit-lower triangular solve, thread-per-column (f32) ----
    if (tid < CH) {
      const int i = tid;
      float u[DTC];
#pragma unroll
      for (int t = 0; t < DTC; ++t) u[t] = uf[t * CH + i];
#pragma unroll
      for (int t = 1; t < DTC; ++t) {
        float acc = u[t];
#pragma unroll
        for (int s = 0; s < DTC - 1; ++s)
          if (s < t) acc += abf[t * DTC + s] * u[s];
        u[t] = acc;
      }
#pragma unroll
      for (int t = 0; t < DTC; ++t) vuB[i * 32 + 16 + t] = (__bf16)u[t];
    }
    __syncthreads();

    // ---- phase 5: y = qk@v + qb@u + wq@S^T, store to global ----
    {
      v8f d = (v8f){0.f, 0.f, 0.f, 0.f, 0.f, 0.f, 0.f, 0.f};
      d = wmma_bf16(ldsA(qkqbA, 32, 0, lane), ldsB(vuB, 32, i0, 0, lane), d); // qk@v + qb@u
      d = wmma_bf16(ldsA(wqA, CH, 0, lane),  ldsB(Sb, CH, i0, 0, lane),  d);
      d = wmma_bf16(ldsA(wqA, CH, 32, lane), ldsB(Sb, CH, i0, 32, lane), d);
#pragma unroll
      for (int r = 0; r < 8; ++r) {
        const int t = r + (hlf << 3);
        gy[tbase + (size_t)t * HC + i0 + n_lo] = d[r];
      }
    }

    // ---- phase 6: state update: S = S*fw[j] + (kwi*fw)^T v + (bwi*fw)^T u ----
    // Wave-private: this wave only touches Sb rows i in its own tile.
#pragma unroll
    for (int jt = 0; jt < 4; ++jt) {
      v8f acc = st[jt];
#pragma unroll
      for (int r = 0; r < 8; ++r) acc[r] *= fwl[(jt << 4) + r + (hlf << 3)];
      acc = wmma_bf16(ldsA(kbT + (jt << 4) * 32, 32, 0, lane),
                      ldsB(vuB, 32, i0, 0, lane), acc);
      st[jt] = acc;
#pragma unroll
      for (int r = 0; r < 8; ++r)
        Sb[(i0 + n_lo) * CH + (jt << 4) + r + (hlf << 3)] = (__bf16)acc[r];
    }
    __syncthreads();
  }
}

extern "C" void kernel_launch(void* const* d_in, const int* in_sizes, int n_in,
                              void* d_out, int out_size, void* d_ws, size_t ws_size,
                              hipStream_t stream) {
  const float* w = (const float*)d_in[0];
  const float* q = (const float*)d_in[1];
  const float* k = (const float*)d_in[2];
  const float* v = (const float*)d_in[3];
  const float* a = (const float*)d_in[4];
  const float* b = (const float*)d_in[5];
  float* y = (float*)d_out;
  rwkv7_heads_kernel<<<dim3(N_HEADS), dim3(128), 0, stream>>>(w, q, k, v, a, b, y);
}
